// GraphConvolution_83288005804153
// MI455X (gfx1250) — matrix operand order repaired and checked by reading, not compiled, not run on hardware
//
#include <hip/hip_runtime.h>
#include <hip/hip_bf16.h>

// ---------------- problem constants (match reference) ----------------
#define N_SRC   50000
#define N_SAMP  50000
#define N_TOTAL 200000
#define NE      1600000
#define DIM     256            // D_IN == D_OUT == 256

#define OUT_ELEMS   ((size_t)N_SAMP * DIM)     // 12.8M floats (out region of d_out)
#define NEWY_ELEMS  ((size_t)N_TOTAL * DIM)    // 51.2M floats

// workspace layout (bytes):
//   [0, 25,600,000)            : feat_bf16  (N_SAMP x DIM ushort)
//   [25,600,000, +131,072)     : Wt_bf16    (DIM x DIM ushort, N-major)
#define WS_FEATBF_OFF 0
#define WS_WT_OFF     ((size_t)N_SAMP * DIM * 2)

typedef __attribute__((ext_vector_type(16))) __bf16 v16bf;
typedef __attribute__((ext_vector_type(8)))  __bf16 v8bf;
typedef __attribute__((ext_vector_type(8)))  float  v8f;

static __device__ __forceinline__ unsigned short f32_to_bf16_rne(float f) {
    unsigned int u = __float_as_uint(f);
    unsigned int lsb = (u >> 16) & 1u;
    u += 0x7fffu + lsb;                 // round to nearest even
    return (unsigned short)(u >> 16);
}

static __device__ __forceinline__ v16bf load_frag16(const unsigned short* p0,
                                                    const unsigned short* p1) {
    v8bf lo = *(const v8bf*)p0;
    v8bf hi = *(const v8bf*)p1;
    return __builtin_shufflevector(lo, hi, 0,1,2,3,4,5,6,7,8,9,10,11,12,13,14,15);
}

// ---------------- kernel 0: zero the feat accumulator ----------------
__global__ __launch_bounds__(256) void k_zero(float4* __restrict__ p) {
    size_t t = (size_t)blockIdx.x * blockDim.x + threadIdx.x;   // OUT_ELEMS/4 threads
    p[t] = make_float4(0.f, 0.f, 0.f, 0.f);
}

// ---------------- kernel 1: edge-parallel SpMM scatter-add ----------------
// thread t: edge e = t>>5, channel block c8 = (t&31)*8 (two float4s)
__global__ __launch_bounds__(256) void k_spmm(const float* __restrict__ x,
                                              const int*   __restrict__ adj_row,
                                              const int*   __restrict__ adj_col,
                                              const float* __restrict__ adj_val,
                                              float* __restrict__ feat) {
    size_t t = (size_t)blockIdx.x * blockDim.x + threadIdx.x;   // NE*32 threads
    size_t e = t >> 5;
    int c = (int)(t & 31) << 3;
    int row = adj_row[e];
    int col = adj_col[e];
    float v  = adj_val[e];
    const float* src = x + (size_t)col * DIM + c;
    const float4 x0 = *(const float4*)(src);
    const float4 x1 = *(const float4*)(src + 4);
    float* dst = feat + (size_t)row * DIM + c;
    unsafeAtomicAdd(dst + 0, v * x0.x);
    unsafeAtomicAdd(dst + 1, v * x0.y);
    unsafeAtomicAdd(dst + 2, v * x0.z);
    unsafeAtomicAdd(dst + 3, v * x0.w);
    unsafeAtomicAdd(dst + 4, v * x1.x);
    unsafeAtomicAdd(dst + 5, v * x1.y);
    unsafeAtomicAdd(dst + 6, v * x1.z);
    unsafeAtomicAdd(dst + 7, v * x1.w);
}

// ---------------- kernel 2: new_y = 0.9 * y_buf (all rows) ----------------
__global__ __launch_bounds__(256) void k_decay(const float4* __restrict__ y,
                                               float4* __restrict__ newy) {
    size_t t = (size_t)blockIdx.x * blockDim.x + threadIdx.x;   // NEWY_ELEMS/4 threads
    float4 v = y[t];
    newy[t] = make_float4(0.9f * v.x, 0.9f * v.y, 0.9f * v.z, 0.9f * v.w);
}

// ---------------- kernel 3: blend + scatter into new_y + bf16 convert ----------------
__global__ __launch_bounds__(256) void k_blend(const float* __restrict__ feat,
                                               const float* __restrict__ y,
                                               const int*   __restrict__ nodes,
                                               float* __restrict__ newy,
                                               unsigned short* __restrict__ featbf) {
    size_t t = (size_t)blockIdx.x * blockDim.x + threadIdx.x;   // N_SAMP*64 threads
    size_t i = t >> 6;
    int c = (int)(t & 63) << 2;
    int node = nodes[i];
    const float4 yv = *(const float4*)(y    + (size_t)node * DIM + c);
    const float4 fv = *(const float4*)(feat + i * (size_t)DIM + c);
    float4 b;
    b.x = 0.9f * yv.x + 0.1f * fv.x;
    b.y = 0.9f * yv.y + 0.1f * fv.y;
    b.z = 0.9f * yv.z + 0.1f * fv.z;
    b.w = 0.9f * yv.w + 0.1f * fv.w;
    *(float4*)(newy + (size_t)node * DIM + c) = b;              // new_y[node] = feat_blend
    ushort4 h;
    h.x = f32_to_bf16_rne(b.x);
    h.y = f32_to_bf16_rne(b.y);
    h.z = f32_to_bf16_rne(b.z);
    h.w = f32_to_bf16_rne(b.w);
    *(ushort4*)(featbf + i * (size_t)DIM + c) = h;              // GEMM A matrix (bf16)
}

// ---------------- kernel 4: W[K][N] -> Wt_bf16[N][K] ----------------
__global__ __launch_bounds__(256) void k_convW(const float* __restrict__ W,
                                               unsigned short* __restrict__ Wt) {
    int t = blockIdx.x * blockDim.x + threadIdx.x;              // DIM*DIM threads
    int k = t & (DIM - 1);
    int n = t >> 8;
    Wt[(size_t)n * DIM + k] = f32_to_bf16_rne(W[(size_t)k * DIM + n]);
}

// ---------------- kernel 5: fused bf16 WMMA GEMM + bias + ELU + row-norm ----------------
// 128 threads = 4 waves per block; block computes a 32 x 256 output tile.
// Each wave: 2 row-tiles x 4 column-tiles -> 8 accumulators; every B fragment
// feeds two WMMAs (reuse across row tiles) so loads are amortized 2x.
__global__ __launch_bounds__(128) void k_gemm(const unsigned short* __restrict__ Abf,
                                              const unsigned short* __restrict__ Wt,
                                              const float* __restrict__ bias,
                                              const float* __restrict__ scale,
                                              const float* __restrict__ offset,
                                              float* __restrict__ out) {
    __shared__ float tile[32][DIM + 16];   // +16 floats pad vs bank conflicts

    const int wave = threadIdx.x >> 5;     // 0..3
    const int lane = threadIdx.x & 31;
    const int lid  = lane & 15;
    const int half = lane >> 4;            // 0: lanes 0-15, 1: lanes 16-31
    const int rowBase = blockIdx.x * 32;

    // CDNA5 16-bit A layout: lane<16 holds K {0..7,16..23}; lane>=16 holds K {8..15,24..31}
    const int koffA = half * 8;
    // CDNA5 16-bit B layout: lane = column N (mod 16); lane<16 holds K {0..7,16..23}; lane>=16 holds K {8..15,24..31}
    const int koffB = half * 8;

    // clamp A-row indices: EXEC must stay all-ones for WMMA, last block has 16 live rows
    const size_t aRow0 = (size_t)min(rowBase +      lid, N_SAMP - 1) * DIM;
    const size_t aRow1 = (size_t)min(rowBase + 16 + lid, N_SAMP - 1) * DIM;

    v8f acc[2][4];
    #pragma unroll
    for (int rt = 0; rt < 2; ++rt)
        #pragma unroll
        for (int j = 0; j < 4; ++j) acc[rt][j] = (v8f){0,0,0,0,0,0,0,0};

    #pragma unroll
    for (int k0 = 0; k0 < DIM; k0 += 32) {
        v16bf a0 = load_frag16(Abf + aRow0 + k0 + koffA, Abf + aRow0 + k0 + koffA + 16);
        v16bf a1 = load_frag16(Abf + aRow1 + k0 + koffA, Abf + aRow1 + k0 + koffA + 16);
        #pragma unroll
        for (int j = 0; j < 4; ++j) {
            const int n = (wave * 4 + j) * 16 + lid;
            const unsigned short* bp = Wt + (size_t)n * DIM + k0 + koffB;
            v16bf b = load_frag16(bp, bp + 16);
            acc[0][j] = __builtin_amdgcn_wmma_f32_16x16x32_bf16(
                            false, a0, false, b, (short)0, acc[0][j], false, false);
            acc[1][j] = __builtin_amdgcn_wmma_f32_16x16x32_bf16(
                            false, a1, false, b, (short)0, acc[1][j], false, false);
        }
    }

    // C/D layout: VGPR r -> M = r + 8*half, N = ntile*16 + lid. Bias + ELU into LDS.
    #pragma unroll
    for (int rt = 0; rt < 2; ++rt) {
        #pragma unroll
        for (int j = 0; j < 4; ++j) {
            const int n = (wave * 4 + j) * 16 + lid;
            const float bn = bias[n];
            #pragma unroll
            for (int r = 0; r < 8; ++r) {
                float v = acc[rt][j][r] + bn;
                v = (v > 0.f) ? v : expm1f(v);
                tile[rt * 16 + r + half * 8][n] = v;
            }
        }
    }
    __syncthreads();

    // row normalization: each wave handles 8 rows; wave32 shfl_xor reduction
    #pragma unroll
    for (int rr = 0; rr < 8; ++rr) {
        const int row  = wave + rr * 4;          // 0..31
        const int grow = rowBase + row;
        float vals[8];
        float sum = 0.f, sq = 0.f;
        #pragma unroll
        for (int j = 0; j < 8; ++j) {
            const int col = lane + j * 32;
            const float v = tile[row][col];
            vals[j] = v; sum += v; sq += v * v;
        }
        #pragma unroll
        for (int m = 16; m >= 1; m >>= 1) {
            sum += __shfl_xor(sum, m, 32);
            sq  += __shfl_xor(sq,  m, 32);
        }
        const float mean = sum * (1.f / DIM);
        const float var  = sq * (1.f / DIM) - mean * mean + 1e-9f;
        const float rs   = rsqrtf(var);
        if (grow < N_SAMP) {
            float* orow = out + (size_t)grow * DIM;
            #pragma unroll
            for (int j = 0; j < 8; ++j) {
                const int col = lane + j * 32;
                orow[col] = (vals[j] - mean) * scale[col] * rs + offset[col];
            }
        }
    }
}

// ---------------- launcher ----------------
extern "C" void kernel_launch(void* const* d_in, const int* in_sizes, int n_in,
                              void* d_out, int out_size, void* d_ws, size_t ws_size,
                              hipStream_t stream) {
    const float* x        = (const float*)d_in[0];
    const int*   adj_row  = (const int*)  d_in[1];
    const int*   adj_col  = (const int*)  d_in[2];
    const float* adj_val  = (const float*)d_in[3];
    const int*   nodes    = (const int*)  d_in[4];
    const float* y_buf    = (const float*)d_in[5];
    const float* W        = (const float*)d_in[6];
    const float* b        = (const float*)d_in[7];
    const float* scale    = (const float*)d_in[8];
    const float* offset   = (const float*)d_in[9];

    float* out_base = (float*)d_out;                 // [0, OUT_ELEMS): final out; feat scratch first
    float* newy     = out_base + OUT_ELEMS;          // [OUT_ELEMS, +NEWY_ELEMS)
    float* feat     = out_base;                      // f32 SpMM accumulator (overwritten by k_gemm)

    unsigned short* featbf = (unsigned short*)((char*)d_ws + WS_FEATBF_OFF);
    unsigned short* Wt     = (unsigned short*)((char*)d_ws + WS_WT_OFF);

    // 0) zero feat accumulator: OUT_ELEMS/4 float4 stores
    k_zero<<<(int)(OUT_ELEMS / 4 / 256), 256, 0, stream>>>((float4*)feat);

    // 1) SpMM scatter-add: NE*32 threads (8 channels per thread)
    k_spmm<<<(int)((size_t)NE * 32 / 256), 256, 0, stream>>>(x, adj_row, adj_col, adj_val, feat);

    // 2) new_y = 0.9 * y_buf: NEWY_ELEMS/4 threads
    k_decay<<<(int)(NEWY_ELEMS / 4 / 256), 256, 0, stream>>>((const float4*)y_buf, (float4*)newy);

    // 3) blend + scatter + bf16 convert: N_SAMP*64 threads
    k_blend<<<(int)((size_t)N_SAMP * 64 / 256), 256, 0, stream>>>(feat, y_buf, nodes, newy, featbf);

    // 4) W -> Wt bf16 (N-major)
    k_convW<<<(DIM * DIM) / 256, 256, 0, stream>>>(W, Wt);

    // 5) fused WMMA GEMM + ELU + row-norm: 32 rows per block
    k_gemm<<<(N_SAMP + 31) / 32, 128, 0, stream>>>(featbf, Wt, b, scale, offset, out_base);
}
